// GPT_87179246174513
// MI455X (gfx1250) — matrix operand order repaired and checked by reading, not compiled, not run on hardware
//
#include <hip/hip_runtime.h>
#include <hip/hip_bf16.h>
#include <math.h>

// ---------------------------------------------------------------------------
// GPT-2 (124M) forward on gfx1250.
//  - All WMMA operands live in HBM as bf16 (weights pre-converted once per
//    launch; activations produced as bf16 by LN / GEMM epilogues / attention).
//  - GEMM: 128x128 block tile, 8 waves, bf16 tiles double-buffered in LDS via
//    GLOBAL_LOAD_ASYNC_TO_LDS_B128 (ASYNCcnt pipelined), zero conversions in
//    the WMMA loop.
//  - Attention: fused causal softmax per 16-row tile, WMMA for QK^T and PV.
// Model constants: V=50304 C=768 L=12 H=12 B=4 T=1024
// ---------------------------------------------------------------------------

#define GPT_V 50304
#define GPT_C 768
#define GPT_L 12
#define GPT_H 12
#define GPT_B 4
#define GPT_T 1024
#define GPT_M (GPT_B * GPT_T)   // 4096 rows
#define GPT_HD 64               // head dim

typedef __bf16 bf16;
typedef bf16  v16bf __attribute__((ext_vector_type(16)));
typedef float v8f   __attribute__((ext_vector_type(8)));
typedef int   v4i   __attribute__((vector_size(16)));   // matches builtin param

#define AS1 __attribute__((address_space(1)))
#define AS3 __attribute__((address_space(3)))

#if defined(__has_builtin)
#  if __has_builtin(__builtin_amdgcn_global_load_async_to_lds_b128)
#    define HAVE_ASYNC_BUILTIN 1
#  endif
#  if __has_builtin(__builtin_amdgcn_s_wait_asynccnt)
#    define HAVE_WAITASYNC_BUILTIN 1
#  endif
#endif

// 16-byte global -> LDS async copy (ASYNCcnt-tracked, bypasses VGPRs)
__device__ __forceinline__ void async_cp16(void* lds, const void* g) {
#ifdef HAVE_ASYNC_BUILTIN
    __builtin_amdgcn_global_load_async_to_lds_b128(
        (AS1 v4i*)(unsigned long long)g,
        (AS3 v4i*)(unsigned)(unsigned long long)lds, 0, 0);
#else
    unsigned l = (unsigned)(unsigned long long)lds;
    asm volatile("global_load_async_to_lds_b128 %0, %1, off"
                 :: "v"(l), "v"(g) : "memory");
#endif
}

template <int N>
__device__ __forceinline__ void wait_asynccnt() {
#ifdef HAVE_WAITASYNC_BUILTIN
    __builtin_amdgcn_s_wait_asynccnt((short)N);
#else
    asm volatile("s_wait_asynccnt %0" :: "i"(N) : "memory");
#endif
}

__device__ __forceinline__ bf16 f2bf(float f) {
    unsigned u = __builtin_bit_cast(unsigned, f);
    u += 0x7FFFu + ((u >> 16) & 1u);              // round-to-nearest-even
    unsigned short h = (unsigned short)(u >> 16);
    return __builtin_bit_cast(bf16, h);
}

__device__ __forceinline__ v8f wmma_bf16(v16bf a, v16bf b, v8f c) {
    // D(16x16,f32) = A(16x32,bf16) * B(32x16,bf16) + C
    return __builtin_amdgcn_wmma_f32_16x16x32_bf16(
        false, a, false, b, (short)0, c, false, false);
}

// A-fragment half-index -> K mapping (16-bit A 16x32, wave32):
// lanes 0-15: halves 0-7 -> K0-7,  halves 8-15 -> K16-23
// lanes 16-31: halves 0-7 -> K8-15, halves 8-15 -> K24-31
__device__ __forceinline__ int kmap(int i, int group) {
    return ((i & 8) << 1) + (group << 3) + (i & 7);
}

__device__ __forceinline__ float gelu_tanh(float u) {
    float c = 0.7978845608028654f * (u + 0.044715f * u * u * u);
    return 0.5f * u * (1.0f + tanhf(c));
}

// ---------------------------------------------------------------------------
// Tiled GEMM:  out[M,N] = A[M,K](bf16) @ W[K,N](bf16) + bias(f32)
// optional fused tanh-GELU; output stored fp32 or bf16 (outBf16 flag).
// Block tile 128x128, 256 threads (8 waves); bf16 tiles double-buffered in
// LDS by async B128 copies (4 per thread per K-step -> s_wait_asynccnt 4).
// A rows padded to 40 halves (80B, 16B-aligned, conflict-free A gather);
// W rows padded to 136 halves (272B, 16B-aligned, <=2-way on B gather).
// Dynamic LDS: 2*128*40*2 + 2*32*136*2 = 37888 B.
// ---------------------------------------------------------------------------
#define AB_STRIDE 40
#define WB_STRIDE 136
#define GEMM_LDS_BYTES (2 * 128 * AB_STRIDE * 2 + 2 * 32 * WB_STRIDE * 2)

__global__ __launch_bounds__(256) void gemm_bf16_kernel(
    const bf16* __restrict__ A, const bf16* __restrict__ W,
    const float* __restrict__ bias, void* __restrict__ out,
    int M, int N, int K, int outBf16, int fuseGelu)
{
    extern __shared__ char smem[];
    bf16* Ab = (bf16*)smem;                              // [2][128][40]
    bf16* Wb = (bf16*)(smem + 2 * 128 * AB_STRIDE * 2);  // [2][32][136]

    const int tid   = threadIdx.x;
    const int lane  = tid & 31;
    const int wave  = tid >> 5;
    const int group = lane >> 4;
    const int mrow  = lane & 15;
    const int n0 = blockIdx.x * 128;
    const int m0 = blockIdx.y * 128;
    const int nk = K >> 5;

    // issue the 4 async B128 copies (per thread) for K-step kt into buffer buf
    auto stage = [&](int kt, int buf) {
        const int k0 = kt * 32;
        // A tile: 128 rows x 32 halves = 512 16B-chunks
#pragma unroll
        for (int it = 0; it < 2; ++it) {
            int cc = tid + it * 256;
            int r = cc >> 2, s = cc & 3;
            async_cp16(&Ab[(buf * 128 + r) * AB_STRIDE + s * 8],
                       A + (size_t)(m0 + r) * K + k0 + s * 8);
        }
        // W tile: 32 rows x 128 halves = 512 16B-chunks
#pragma unroll
        for (int it = 0; it < 2; ++it) {
            int cc = tid + it * 256;
            int r = cc >> 4, s = cc & 15;
            async_cp16(&Wb[(buf * 32 + r) * WB_STRIDE + s * 8],
                       W + (size_t)(k0 + r) * N + n0 + s * 8);
        }
    };

    v8f acc[8];
#pragma unroll
    for (int i = 0; i < 8; ++i) acc[i] = {};

    stage(0, 0);                                  // prologue: fill buffer 0

    for (int kt = 0; kt < nk; ++kt) {
        const int buf = kt & 1;
        if (kt + 1 < nk) {
            stage(kt + 1, buf ^ 1);               // prefetch next stage
            wait_asynccnt<4>();                   // stage kt landed (in-order)
        } else {
            wait_asynccnt<0>();
        }
        __syncthreads();

        // ---- A fragment: contiguous bf16 octets -> ds_load_b64 x4 ----
        v16bf a;
#pragma unroll
        for (int i = 0; i < 16; ++i)
            a[i] = Ab[(buf * 128 + wave * 16 + mrow) * AB_STRIDE + kmap(i, group)];

        // ---- 8 WMMAs across the 128 N columns (B: lane->N, half->K via kmap) ----
#pragma unroll
        for (int nt = 0; nt < 8; ++nt) {
            v16bf b;
#pragma unroll
            for (int i = 0; i < 16; ++i)
                b[i] = Wb[(buf * 32 + kmap(i, group)) * WB_STRIDE + nt * 16 + mrow];
            acc[nt] = wmma_bf16(a, b, acc[nt]);
        }
        __syncthreads();                          // protect buf before reuse
    }

    // ---- epilogue: C layout = (VGPR r, lane) -> (M = r+8*group, N = lane&15)
#pragma unroll
    for (int nt = 0; nt < 8; ++nt) {
        int col = n0 + nt * 16 + mrow;
        float bv = bias ? bias[col] : 0.0f;
#pragma unroll
        for (int r = 0; r < 8; ++r) {
            int row = m0 + wave * 16 + r + 8 * group;
            float v = acc[nt][r] + bv;
            if (fuseGelu) v = gelu_tanh(v);
            if (outBf16) ((bf16*)out)[(size_t)row * N + col] = f2bf(v);
            else         ((float*)out)[(size_t)row * N + col] = v;
        }
    }
}

// ---------------------------------------------------------------------------
// Fused causal attention for one (b, h, 16-query-row tile); 1 wave per block.
// qkv is bf16 [B,T,3C] (q +0, k +C, v +2C). Scores UNSCALED (faithful).
// K tiles staged through a small LDS tile (coalesced uint4 copies) so the
// per-feature gather hits LDS, not global. Two-pass softmax over a 16x1024
// fp32 LDS strip; P@V via WMMA with global V loads (lane->feature, stride 3C per key).
// Dynamic LDS: 64KB scores + 16*72*2 B K-tile = 67840 B.
// ---------------------------------------------------------------------------
#define KS_STRIDE 72
#define ATTN_LDS_BYTES (16 * GPT_T * 4 + 16 * KS_STRIDE * 2)

__global__ __launch_bounds__(32) void attn_kernel(
    const bf16* __restrict__ qkv, bf16* __restrict__ att)
{
    extern __shared__ char smem[];
    float* sc = (float*)smem;                        // [16][1024] scores
    bf16*  Ks = (bf16*)(smem + 16 * GPT_T * 4);      // [16][72] K tile
    const int itile = blockIdx.x;                    // 0..63
    const int hh    = blockIdx.y;                    // 0..11
    const int bb    = blockIdx.z;                    // 0..3
    const int lane  = threadIdx.x;
    const int group = lane >> 4;
    const int mrow  = lane & 15;
    const int i0    = itile * 16;
    const int C3    = 3 * GPT_C;
    const size_t base = (size_t)bb * GPT_T * C3;
    const int qoff = hh * GPT_HD;
    const int koff = GPT_C + hh * GPT_HD;
    const int voff = 2 * GPT_C + hh * GPT_HD;

    // Q fragments: per-lane contiguous bf16 octets (global b128 loads)
    v16bf aq0, aq1;
#pragma unroll
    for (int i = 0; i < 16; ++i) {
        int kk = kmap(i, group);
        const bf16* qrow = qkv + base + (size_t)(i0 + mrow) * C3 + qoff;
        aq0[i] = qrow[kk];
        aq1[i] = qrow[32 + kk];
    }

    // ---- pass A: scores S = Q K^T for all causal key tiles ----
    for (int jt = 0; jt <= itile; ++jt) {
        int j0 = jt * 16;
        // stage K tile [16 keys x 64 features] bf16 -> LDS (128 16B-chunks)
#pragma unroll
        for (int it = 0; it < 4; ++it) {
            int cc = lane + it * 32;
            int r = cc >> 3, s = cc & 7;
            *(uint4*)&Ks[r * KS_STRIDE + s * 8] =
                *(const uint4*)(qkv + base + (size_t)(j0 + r) * C3 + koff + s * 8);
        }
        v16bf bk0, bk1;                            // B: lane->N(key), half->K(feature)
#pragma unroll
        for (int i = 0; i < 16; ++i) {
            bk0[i] = Ks[mrow * KS_STRIDE + kmap(i, group)];
            bk1[i] = Ks[mrow * KS_STRIDE + 32 + kmap(i, group)];
        }
        v8f s = {};
        s = wmma_bf16(aq0, bk0, s);
        s = wmma_bf16(aq1, bk1, s);
#pragma unroll
        for (int r = 0; r < 8; ++r) {
            int row  = r + 8 * group;
            int gcol = j0 + mrow;
            float v = s[r];
            if (jt == itile && gcol > i0 + row) v = -3.0e38f;   // causal mask
            sc[row * GPT_T + gcol] = v;
        }
    }
    __syncthreads();

    // ---- softmax per row (threads 0..15 each own a row) ----
    const int ncols = (itile + 1) * 16;
    const int pad   = (ncols + 31) & ~31;          // zero-pad to K=32 multiple
    if (lane < 16) {
        float* rowp = sc + lane * GPT_T;
        float mx = -3.0e38f;
        for (int j = 0; j < ncols; ++j) mx = fmaxf(mx, rowp[j]);
        float sum = 0.0f;
        for (int j = 0; j < ncols; ++j) { float e = __expf(rowp[j] - mx); rowp[j] = e; sum += e; }
        float inv = 1.0f / sum;
        for (int j = 0; j < ncols; ++j) rowp[j] *= inv;
        for (int j = ncols; j < pad; ++j) rowp[j] = 0.0f;
    }
    __syncthreads();

    // ---- pass B: O = P V  (K runs over keys in chunks of 32) ----
    v8f o[4];
#pragma unroll
    for (int nt = 0; nt < 4; ++nt) o[nt] = {};

    for (int ch = 0; ch < pad; ch += 32) {
        v16bf ap;                                   // A: lane->M(query), half->K(key)
#pragma unroll
        for (int i = 0; i < 16; ++i)
            ap[i] = f2bf(sc[mrow * GPT_T + ch + kmap(i, group)]);
        const bf16* vrow = qkv + base + voff + mrow;
#pragma unroll
        for (int nt = 0; nt < 4; ++nt) {
            v16bf bv;                               // B: lane->N(feature), half->K(key)
#pragma unroll
            for (int i = 0; i < 16; ++i) bv[i] = vrow[(size_t)(ch + kmap(i, group)) * C3 + nt * 16];
            o[nt] = wmma_bf16(ap, bv, o[nt]);
        }
    }

    // ---- write [16 x 64] head output (bf16) into att[B,T,C] ----
#pragma unroll
    for (int nt = 0; nt < 4; ++nt) {
        int c = hh * GPT_HD + nt * 16 + mrow;
#pragma unroll
        for (int r = 0; r < 8; ++r) {
            int row = i0 + r + 8 * group;
            att[((size_t)bb * GPT_T + row) * GPT_C + c] = f2bf(o[nt][r]);
        }
    }
}

// ---------------------------------------------------------------------------
// Weight conversion kernels (once per launch; ~0.5 GB total = ~25 us at HBM)
// ---------------------------------------------------------------------------
__global__ void convert_bf16_kernel(const float* __restrict__ in,
                                    bf16* __restrict__ out, int n)
{
    int i = (blockIdx.x * 256 + threadIdx.x) * 4;
    if (i + 3 < n) {
        float4 f = *(const float4*)(in + i);
        out[i + 0] = f2bf(f.x);
        out[i + 1] = f2bf(f.y);
        out[i + 2] = f2bf(f.z);
        out[i + 3] = f2bf(f.w);
    }
}

// wte [V,C] fp32 -> wteT [C,V] bf16 via 32x32 LDS tile transpose
__global__ __launch_bounds__(256) void transpose_bf16_kernel(
    const float* __restrict__ wte, bf16* __restrict__ wteT)
{
    __shared__ float tile[32][33];
    const int n0 = blockIdx.x * 32;     // vocab
    const int k0 = blockIdx.y * 32;     // channel
    const int r  = threadIdx.x >> 3;
    const int c4 = (threadIdx.x & 7) * 4;

    float4 f = *(const float4*)(wte + (size_t)(n0 + r) * GPT_C + k0 + c4);
    tile[r][c4 + 0] = f.x; tile[r][c4 + 1] = f.y;
    tile[r][c4 + 2] = f.z; tile[r][c4 + 3] = f.w;
    __syncthreads();

#pragma unroll
    for (int j = 0; j < 4; ++j)
        wteT[(size_t)(k0 + r) * GPT_V + n0 + c4 + j] = f2bf(tile[c4 + j][r]);
}

// ---------------------------------------------------------------------------
// Elementwise / reduction kernels
// ---------------------------------------------------------------------------
__global__ void embed_kernel(const int* __restrict__ x,
                             const float* __restrict__ wte,
                             const float* __restrict__ wpe,
                             float* __restrict__ h)
{
    int row = blockIdx.x;                 // 0..4095  (b*T + t)
    int t   = row & (GPT_T - 1);
    int tok = x[row];
    for (int c = threadIdx.x; c < GPT_C; c += 256)
        h[(size_t)row * GPT_C + c] =
            wte[(size_t)tok * GPT_C + c] + wpe[(size_t)t * GPT_C + c];
}

// LayerNorm: fp32 in, bf16 out (always feeds a GEMM A-input)
__global__ __launch_bounds__(256) void ln_kernel(
    const float* __restrict__ x, const float* __restrict__ w,
    const float* __restrict__ b, bf16* __restrict__ out)
{
    __shared__ float red[256];
    const int row = blockIdx.x;
    const float* xr = x + (size_t)row * GPT_C;

    float s = 0.0f;
    for (int c = threadIdx.x; c < GPT_C; c += 256) s += xr[c];
    red[threadIdx.x] = s; __syncthreads();
    for (int st = 128; st > 0; st >>= 1) {
        if (threadIdx.x < st) red[threadIdx.x] += red[threadIdx.x + st];
        __syncthreads();
    }
    float mean = red[0] / GPT_C;
    __syncthreads();

    float s2 = 0.0f;
    for (int c = threadIdx.x; c < GPT_C; c += 256) { float d = xr[c] - mean; s2 += d * d; }
    red[threadIdx.x] = s2; __syncthreads();
    for (int st = 128; st > 0; st >>= 1) {
        if (threadIdx.x < st) red[threadIdx.x] += red[threadIdx.x + st];
        __syncthreads();
    }
    float rstd = rsqrtf(red[0] / GPT_C + 1e-5f);

    bf16* orow = out + (size_t)row * GPT_C;
    for (int c = threadIdx.x; c < GPT_C; c += 256)
        orow[c] = f2bf((xr[c] - mean) * rstd * w[c] + b[c]);
}

__global__ void add_kernel(float* __restrict__ a, const float* __restrict__ b, int n)
{
    int i = blockIdx.x * 256 + threadIdx.x;
    if (i < n) a[i] += b[i];
}

// per-row  (logsumexp - logit[target]) ; deterministic final reduce below
__global__ __launch_bounds__(256) void loss_row_kernel(
    const float* __restrict__ logits, const int* __restrict__ targets,
    float* __restrict__ row_loss)
{
    __shared__ float red[256];
    const int row = blockIdx.x;
    const float* lr = logits + (size_t)row * GPT_V;

    float mx = -3.0e38f;
    for (int j = threadIdx.x; j < GPT_V; j += 256) mx = fmaxf(mx, lr[j]);
    red[threadIdx.x] = mx; __syncthreads();
    for (int st = 128; st > 0; st >>= 1) {
        if (threadIdx.x < st) red[threadIdx.x] = fmaxf(red[threadIdx.x], red[threadIdx.x + st]);
        __syncthreads();
    }
    mx = red[0]; __syncthreads();

    float s = 0.0f;
    for (int j = threadIdx.x; j < GPT_V; j += 256) s += __expf(lr[j] - mx);
    red[threadIdx.x] = s; __syncthreads();
    for (int st = 128; st > 0; st >>= 1) {
        if (threadIdx.x < st) red[threadIdx.x] += red[threadIdx.x + st];
        __syncthreads();
    }
    if (threadIdx.x == 0) {
        float lse = mx + logf(red[0]);
        row_loss[row] = lse - lr[targets[row]];
    }
}

__global__ __launch_bounds__(256) void loss_reduce_kernel(
    const float* __restrict__ row_loss, float* __restrict__ loss_out)
{
    __shared__ float red[256];
    float s = 0.0f;
    for (int i = threadIdx.x; i < GPT_M; i += 256) s += row_loss[i];
    red[threadIdx.x] = s; __syncthreads();
    for (int st = 128; st > 0; st >>= 1) {
        if (threadIdx.x < st) red[threadIdx.x] += red[threadIdx.x + st];
        __syncthreads();
    }
    if (threadIdx.x == 0) loss_out[0] = red[0] / (float)GPT_M;
}

// ---------------------------------------------------------------------------
// Host orchestration
// ---------------------------------------------------------------------------
extern "C" void kernel_launch(void* const* d_in, const int* in_sizes, int n_in,
                              void* d_out, int out_size, void* d_ws, size_t ws_size,
                              hipStream_t stream)
{
    (void)in_sizes; (void)n_in; (void)out_size; (void)ws_size;

    const int*   x       = (const int*)  d_in[0];
    const int*   targets = (const int*)  d_in[1];
    const float* wte     = (const float*)d_in[2];
    const float* wpe     = (const float*)d_in[3];
    const float* ln1_w   = (const float*)d_in[4];
    const float* ln1_b   = (const float*)d_in[5];
    const float* attn_w  = (const float*)d_in[6];
    const float* attn_b  = (const float*)d_in[7];
    const float* proj_w  = (const float*)d_in[8];
    const float* proj_b  = (const float*)d_in[9];
    const float* ln2_w   = (const float*)d_in[10];
    const float* ln2_b   = (const float*)d_in[11];
    const float* fc_w    = (const float*)d_in[12];
    const float* fc_b    = (const float*)d_in[13];
    const float* fc2_w   = (const float*)d_in[14];
    const float* fc2_b   = (const float*)d_in[15];
    const float* lnf_w   = (const float*)d_in[16];
    const float* lnf_b   = (const float*)d_in[17];

    float* logits = (float*)d_out;                       // [M, V] fp32
    float* loss   = logits + (size_t)GPT_M * GPT_V;      // single float

    // workspace carve-out (~330 MB: bf16 weights 247 MB + activations)
    size_t off = 0;
    auto carveB = [&](size_t nbytes) {
        char* p = (char*)d_ws + off;
        off += (nbytes + 255) & ~(size_t)255;
        return p;
    };
    const int M = GPT_M, C = GPT_C, C3 = 3 * GPT_C, C4 = 4 * GPT_C;
    const int nQKV = GPT_L * C * C3, nPROJ = GPT_L * C * C,
              nFC = GPT_L * C * C4, nFC2 = GPT_L * C4 * C;

    bf16* wqkv_bf = (bf16*)carveB((size_t)nQKV * 2);
    bf16* wproj_bf= (bf16*)carveB((size_t)nPROJ * 2);
    bf16* wfc_bf  = (bf16*)carveB((size_t)nFC * 2);
    bf16* wfc2_bf = (bf16*)carveB((size_t)nFC2 * 2);
    bf16* wteT_bf = (bf16*)carveB((size_t)C * GPT_V * 2);    // [C,V]
    float* h_buf  = (float*)carveB((size_t)M * C * 4);       // residual (fp32)
    float* tmp    = (float*)carveB((size_t)M * C * 4);       // proj/fc2 out (fp32)
    bf16* hn_bf   = (bf16*)carveB((size_t)M * C * 2);        // LN output
    bf16* qkv_bf  = (bf16*)carveB((size_t)M * C3 * 2);       // qkv
    bf16* att_bf  = (bf16*)carveB((size_t)M * C * 2);        // attention out
    bf16* fca_bf  = (bf16*)carveB((size_t)M * C4 * 2);       // gelu(fc) out

    const int nElemC = M * C;
    const dim3 blk256(256), blk32(32);
    const size_t gemmLds = GEMM_LDS_BYTES;
    const size_t attnLds = ATTN_LDS_BYTES;

    // ---- one-time (per launch) weight conversion to bf16 ----
    convert_bf16_kernel<<<(nQKV / 4 + 255) / 256, blk256, 0, stream>>>(attn_w, wqkv_bf, nQKV);
    convert_bf16_kernel<<<(nPROJ / 4 + 255) / 256, blk256, 0, stream>>>(proj_w, wproj_bf, nPROJ);
    convert_bf16_kernel<<<(nFC / 4 + 255) / 256, blk256, 0, stream>>>(fc_w, wfc_bf, nFC);
    convert_bf16_kernel<<<(nFC2 / 4 + 255) / 256, blk256, 0, stream>>>(fc2_w, wfc2_bf, nFC2);
    transpose_bf16_kernel<<<dim3(GPT_V / 32, C / 32), blk256, 0, stream>>>(wte, wteT_bf);

    // token + position embedding (fp32 residual stream)
    embed_kernel<<<M, blk256, 0, stream>>>(x, wte, wpe, h_buf);

    for (int l = 0; l < GPT_L; ++l) {
        // hn = LN1(h)  (bf16)
        ln_kernel<<<M, blk256, 0, stream>>>(h_buf, ln1_w + l * C, ln1_b + l * C, hn_bf);
        // qkv = hn @ attn_w[l] + attn_b[l]  (bf16 out)
        gemm_bf16_kernel<<<dim3(C3 / 128, M / 128), blk256, gemmLds, stream>>>(
            hn_bf, wqkv_bf + (size_t)l * C * C3, attn_b + (size_t)l * C3,
            qkv_bf, M, C3, C, 1, 0);
        // fused causal attention (unscaled scores), bf16 out
        attn_kernel<<<dim3(GPT_T / 16, GPT_H, GPT_B), blk32, attnLds, stream>>>(
            qkv_bf, att_bf);
        // h += att @ proj_w[l] + proj_b[l]  (fp32 out)
        gemm_bf16_kernel<<<dim3(C / 128, M / 128), blk256, gemmLds, stream>>>(
            att_bf, wproj_bf + (size_t)l * C * C, proj_b + (size_t)l * C,
            tmp, M, C, C, 0, 0);
        add_kernel<<<(nElemC + 255) / 256, blk256, 0, stream>>>(h_buf, tmp, nElemC);
        // hn = LN2(h)
        ln_kernel<<<M, blk256, 0, stream>>>(h_buf, ln2_w + l * C, ln2_b + l * C, hn_bf);
        // fcact = gelu(hn @ fc_w[l] + fc_b[l])  (fused gelu, bf16 out)
        gemm_bf16_kernel<<<dim3(C4 / 128, M / 128), blk256, gemmLds, stream>>>(
            hn_bf, wfc_bf + (size_t)l * C * C4, fc_b + (size_t)l * C4,
            fca_bf, M, C4, C, 1, 1);
        // h += fcact @ fc2_w[l] + fc2_b[l]  (fp32 out)
        gemm_bf16_kernel<<<dim3(C / 128, M / 128), blk256, gemmLds, stream>>>(
            fca_bf, wfc2_bf + (size_t)l * C4 * C, fc2_b + (size_t)l * C,
            tmp, M, C, C4, 0, 0);
        add_kernel<<<(nElemC + 255) / 256, blk256, 0, stream>>>(h_buf, tmp, nElemC);
    }

    // final LN + tied LM head: logits = LN_f(h) @ wteT  (fp32 out to d_out)
    ln_kernel<<<M, blk256, 0, stream>>>(h_buf, lnf_w, lnf_b, hn_bf);
    gemm_bf16_kernel<<<dim3(GPT_V / 128, M / 128), blk256, gemmLds, stream>>>(
        hn_bf, wteT_bf, nullptr, logits, M, GPT_V, C, 0, 0);

    // loss = mean over rows of (logsumexp - logit[target]); deterministic reduce
    loss_row_kernel<<<M, blk256, 0, stream>>>(logits, targets, tmp);
    loss_reduce_kernel<<<1, blk256, 0, stream>>>(tmp, loss);
}
